// WindowAttention_16226386444489
// MI455X (gfx1250) — matrix operand order repaired and checked by reading, not compiled, not run on hardware
//
#include <hip/hip_runtime.h>

typedef _Float16 f16;
typedef __attribute__((ext_vector_type(16))) _Float16 v16h;
typedef __attribute__((ext_vector_type(8)))  _Float16 v8h;
typedef __attribute__((ext_vector_type(8)))  float    v8f;

#define WMMA_F32_F16(a,b,c) \
  __builtin_amdgcn_wmma_f32_16x16x32_f16(false,(a),false,(b),(short)0,(c),false,false)

// ---- CDNA5 async global->LDS copy (ASYNCcnt-tracked, bypasses VGPR round trip) ----
__device__ __forceinline__ void async_g2l_b128(void* lds_dst, const void* gsrc) {
  // low 32 bits of a generic pointer into __shared__ = LDS byte offset
  unsigned int lds = (unsigned int)(unsigned long long)lds_dst;
  asm volatile("global_load_async_to_lds_b128 %0, %1, off"
               :: "v"(lds), "v"(gsrc) : "memory");
}
__device__ __forceinline__ void wait_asynccnt0() {
  asm volatile("s_wait_asynccnt 0x0" ::: "memory");
}

// ---- fragment loaders (CDNA5 wave32 layouts, 16-bit A 16x32 / B 32x16) ----
// A: lanes 0-15 hold rows M=0..15 with K = {hi*8..hi*8+7} U {16+hi*8..}, hi = lane>>4
__device__ __forceinline__ v16h load_a16(const f16* __restrict__ base, int row0, int k0,
                                         int ld, int lane) {
  const int r = lane & 15, hi = lane >> 4;
  const f16* p = base + (size_t)(row0 + r) * ld + k0 + hi * 8;
  v8h lo = *(const v8h*)p;
  v8h hh = *(const v8h*)(p + 16);
  v16h a;
#pragma unroll
  for (int i = 0; i < 8; ++i) { a[i] = lo[i]; a[8 + i] = hh[i]; }
  return a;
}
// B from transposed storage BT[n][k] (so the 16 K-values per lane are contiguous):
// lane holds column N = lane&15, K = {hi*8..hi*8+7} U {16+hi*8..16+hi*8+7}
__device__ __forceinline__ v16h load_bT(const f16* __restrict__ baseT, int col0, int k0,
                                        int ld, int lane) {
  const int n = lane & 15, hi = lane >> 4;
  const f16* p = baseT + (size_t)(col0 + n) * ld + k0 + hi * 8;
  v8h x0 = *(const v8h*)p;
  v8h x1 = *(const v8h*)(p + 16);
  v16h b;
#pragma unroll
  for (int i = 0; i < 8; ++i) { b[i] = x0[i]; b[8 + i] = x1[i]; }
  return b;
}

// ---- LDS layout (≈119.3 KB -> 2 workgroups per WGP of the 320 KB pool) ----
struct __align__(16) SMem {
  f16   w[27648];    // WqkvT (288x96). After phase A: [0..4095]=attn probs, [4096..13311]=WoutT
  f16   x[6144];     // x window f16 (64x96); reused as attn_out (64x96)
  f16   qkv[12288];  // [tok 64][ Q(scaled) 96 | K 96 ]  (ld = 192)
  f16   vT[6144];    // V transposed: [d 96][tok 64]
  float scores[4096];// per-head 64x64 scores (f32)
  float pos[343];    // 7x7x7 relative position table
};

__global__ void prep_weights(const float* __restrict__ w_qkv, const float* __restrict__ w_out,
                             f16* __restrict__ wqkvT, f16* __restrict__ woutT) {
  int i = blockIdx.x * 256 + threadIdx.x;
  if (i < 96 * 288) { int k = i / 288, n = i % 288; wqkvT[n * 96 + k] = (f16)w_qkv[i]; }
  if (i < 96 * 96)  { int k = i / 96,  n = i % 96;  woutT[n * 96 + k] = (f16)w_out[i]; }
}

__global__ __launch_bounds__(256, 2)
void win_attn_fused(const float* __restrict__ x, const float* __restrict__ pos,
                    const float* __restrict__ b_out,
                    const f16* __restrict__ wqkvT, const f16* __restrict__ woutT,
                    float* __restrict__ out) {
  __shared__ SMem sm;
  const int tid = threadIdx.x;
  const int wave = tid >> 5, lane = tid & 31;
  const int wid = blockIdx.x;
  const int bb = wid >> 12;
  const int w3 = wid & 4095;
  const int wx = w3 >> 8, wy = (w3 >> 4) & 15, wz = w3 & 15;

  // ---- Phase 0: async-stage WqkvT to LDS; stage x window (f32->f16) ----
  // WqkvT: 3456 x b128, WoutT: 1152 x b128 (goes to its final spot sm.w+4096 only
  // after phase A, so stage it into sm.scores temporarily? No -- WoutT region
  // overlaps live WqkvT; stage WoutT after phase A with async too.)
  for (int idx = tid; idx < 3456; idx += 256)
    async_g2l_b128(&sm.w[idx * 8], wqkvT + idx * 8);
  for (int idx = tid; idx < 1536; idx += 256) {          // 64 tok * 24 float4
    int tok = idx / 24, c4 = idx % 24;
    int tx = tok >> 4, ty = (tok >> 2) & 3, tz = tok & 3;
    size_t row = ((((size_t)bb * 64 + wx * 4 + tx) * 64 + wy * 4 + ty) * 64 + wz * 4 + tz) * 96;
    const float4 v = *(const float4*)(x + row + c4 * 4);
    f16* d = &sm.x[tok * 96 + c4 * 4];
    d[0] = (f16)v.x; d[1] = (f16)v.y; d[2] = (f16)v.z; d[3] = (f16)v.w;
  }
  for (int idx = tid; idx < 343; idx += 256) sm.pos[idx] = pos[idx];
  wait_asynccnt0();
  __syncthreads();

  // ---- Phase A: QKV(64x288) = Xw(64x96) * Wqkv(96x288), 72 tiles, 9/wave ----
  const float scale = 0.17677669529663687f;              // 32^-0.5 folded into Q
  for (int t = wave; t < 72; t += 8) {
    int m = t / 18, n = t % 18;
    v8f acc = {};
#pragma unroll
    for (int kk = 0; kk < 3; ++kk) {
      v16h a = load_a16(sm.x, m * 16, kk * 32, 96, lane);
      v16h b = load_bT(sm.w, n * 16, kk * 32, 96, lane);
      acc = WMMA_F32_F16(a, b, acc);
    }
    const int col = lane & 15, hi = lane >> 4;
    if (n < 6) {                                         // Q (scaled)
#pragma unroll
      for (int r = 0; r < 8; ++r)
        sm.qkv[(m * 16 + r + 8 * hi) * 192 + n * 16 + col] = (f16)(acc[r] * scale);
    } else if (n < 12) {                                 // K
#pragma unroll
      for (int r = 0; r < 8; ++r)
        sm.qkv[(m * 16 + r + 8 * hi) * 192 + n * 16 + col] = (f16)acc[r];
    } else {                                             // V -> transposed [d][tok]
#pragma unroll
      for (int r = 0; r < 8; ++r)
        sm.vT[((n - 12) * 16 + col) * 64 + m * 16 + r + 8 * hi] = (f16)acc[r];
    }
  }
  __syncthreads();

  // Async-stage WoutT into freed region of sm.w. Its first consumer (phase E) is
  // behind several barriers; the asynccnt wait below the head loop is not needed
  // because we wait before the first barrier that precedes phase E.
  for (int idx = tid; idx < 1152; idx += 256)
    async_g2l_b128(&sm.w[4096 + idx * 8], woutT + idx * 8);
  f16* s_attn = sm.w;                                    // 64x64 f16 probs
  const f16* s_woutT = sm.w + 4096;                      // 96x96 f16 (BT layout)

  // ---- per-head attention ----
  for (int h = 0; h < 3; ++h) {
    // scores(64x64) = Qh(64x32) * Kh^T(32x64) + bias ; 16 tiles, one K-step each
    for (int t = wave; t < 16; t += 8) {
      int mi = t >> 2, ni = t & 3;
      v16h a = load_a16(sm.qkv + h * 32, mi * 16, 0, 192, lane);
      v16h b = load_bT(sm.qkv + 96 + h * 32, ni * 16, 0, 192, lane);
      v8f acc = {};
      acc = WMMA_F32_F16(a, b, acc);
      const int col = lane & 15, hi = lane >> 4;
      const int tj = ni * 16 + col;
      const int jx = tj >> 4, jy = (tj >> 2) & 3, jz = tj & 3;
#pragma unroll
      for (int r = 0; r < 8; ++r) {
        int ti = mi * 16 + r + 8 * hi;
        int ix = ti >> 4, iy = (ti >> 2) & 3, iz = ti & 3;
        float bias = sm.pos[((jx - ix + 3) * 7 + (jy - iy + 3)) * 7 + (jz - iz + 3)];
        sm.scores[ti * 64 + tj] = acc[r] + bias;
      }
    }
    __syncthreads();

    // f32 softmax: 4 lanes per row (all 256 threads), shfl_xor reduction over the
    // 4-lane group, probs written back as two ds_store_b128 per thread.
    {
      const int row = tid >> 2, seg = tid & 3;
      const float* sr = sm.scores + row * 64 + seg * 16;
      float e[16];
#pragma unroll
      for (int j = 0; j < 16; ++j) e[j] = sr[j];
      float mx = e[0];
#pragma unroll
      for (int j = 1; j < 16; ++j) mx = fmaxf(mx, e[j]);
      mx = fmaxf(mx, __shfl_xor(mx, 1));
      mx = fmaxf(mx, __shfl_xor(mx, 2));
      float s = 0.f;
#pragma unroll
      for (int j = 0; j < 16; ++j) { e[j] = __expf(e[j] - mx); s += e[j]; }
      s += __shfl_xor(s, 1);
      s += __shfl_xor(s, 2);
      const float inv = 1.f / s;
      v8h h0, h1;
#pragma unroll
      for (int j = 0; j < 8; ++j) { h0[j] = (f16)(e[j] * inv); h1[j] = (f16)(e[8 + j] * inv); }
      f16* arow = s_attn + row * 64 + seg * 16;
      *(v8h*)arow = h0;
      *(v8h*)(arow + 8) = h1;
    }
    __syncthreads();

    // attn_out_h(64x32) = P(64x64) * Vh(64x32) ; 8 tiles, 1/wave
    {
      const int mi = wave >> 1, nd = wave & 1;
      v8f acc = {};
#pragma unroll
      for (int ks = 0; ks < 2; ++ks) {
        v16h a = load_a16(s_attn, mi * 16, ks * 32, 64, lane);
        v16h b = load_bT(sm.vT + h * 32 * 64, nd * 16, ks * 32, 64, lane);
        acc = WMMA_F32_F16(a, b, acc);
      }
      const int col = lane & 15, hi = lane >> 4;
#pragma unroll
      for (int r = 0; r < 8; ++r)
        sm.x[(mi * 16 + r + 8 * hi) * 96 + h * 32 + nd * 16 + col] = (f16)acc[r];
    }
    if (h == 0) wait_asynccnt0();    // WoutT staging complete before next barrier
    __syncthreads();
  }

  // ---- Phase E: out(64x96) = AO(64x96) * Wout(96x96) + b_out ; 24 tiles, 3/wave ----
  for (int t = wave; t < 24; t += 8) {
    int m = t / 6, n = t % 6;
    v8f acc = {};
#pragma unroll
    for (int kk = 0; kk < 3; ++kk) {
      v16h a = load_a16(sm.x, m * 16, kk * 32, 96, lane);
      v16h b = load_bT(s_woutT, n * 16, kk * 32, 96, lane);
      acc = WMMA_F32_F16(a, b, acc);
    }
    const int col = lane & 15, hi = lane >> 4;
    const float bv = b_out[n * 16 + col];
#pragma unroll
    for (int r = 0; r < 8; ++r) {
      int tok = m * 16 + r + 8 * hi;
      int tx = tok >> 4, ty = (tok >> 2) & 3, tz = tok & 3;
      size_t row = ((((size_t)bb * 64 + wx * 4 + tx) * 64 + wy * 4 + ty) * 64 + wz * 4 + tz) * 96;
      out[row + n * 16 + col] = acc[r] + bv;
    }
  }
}

extern "C" void kernel_launch(void* const* d_in, const int* in_sizes, int n_in,
                              void* d_out, int out_size, void* d_ws, size_t ws_size,
                              hipStream_t stream) {
  const float* x     = (const float*)d_in[0];
  const float* w_qkv = (const float*)d_in[1];
  const float* pos   = (const float*)d_in[2];
  const float* w_out = (const float*)d_in[3];
  const float* b_out = (const float*)d_in[4];
  float* out = (float*)d_out;

  f16* wqkvT = (f16*)d_ws;                               // 27648 halfs
  f16* woutT = (f16*)((char*)d_ws + 96 * 288 * sizeof(f16)); // 9216 halfs

  prep_weights<<<(96 * 288 + 255) / 256, 256, 0, stream>>>(w_qkv, w_out, wqkvT, woutT);
  win_attn_fused<<<2 * 16 * 16 * 16, 256, 0, stream>>>(x, pos, b_out, wqkvT, woutT, out);
}
